// RNNSystem_37108517437490
// MI455X (gfx1250) — matrix operand order repaired and checked by reading, not compiled, not run on hardware
//
#include <hip/hip_runtime.h>

typedef __attribute__((ext_vector_type(16))) _Float16 v16h;
typedef __attribute__((ext_vector_type(8)))  float    v8f;
typedef __attribute__((ext_vector_type(2)))  _Float16 h2v;

#define S_LEN 4096
#define B_TOT 8192
#define TILES_PER_BLOCK 4
#define BLOCK_THREADS (TILES_PER_BLOCK * 32)

// tanh(v) = 1 - 2/(exp(2v)+1), exp via v_exp_f32 (exp2), rcp via v_rcp_f32.
// Saturates correctly at +/-1 for large |v| (inf -> rcp 0; 0 -> 1-2 = -1).
__device__ __forceinline__ float fast_tanh(float v) {
    float e = __builtin_amdgcn_exp2f(v * 2.8853900817779268f); // 2*log2(e)
    float r = __builtin_amdgcn_rcpf(e + 1.0f);
    return __builtin_fmaf(-2.0f, r, 1.0f);
}

// Swap lanes L <-> L^16 (SWAPX16): group-of-32 swizzle, xor=0x10, and=0x1f.
__device__ __forceinline__ float swap16(float v) {
    return __builtin_bit_cast(float,
        __builtin_amdgcn_ds_swizzle(__builtin_bit_cast(int, v), 0x401F));
}

// Pack two f32 -> packed 2xf16 (v_cvt_pk_rtz_f16_f32), as our h2v type.
__device__ __forceinline__ h2v pack_f16(float a, float b) {
    return __builtin_bit_cast(h2v, __builtin_amdgcn_cvt_pkrtz(a, b));
}

__global__ __launch_bounds__(BLOCK_THREADS) void rnn_wmma_kernel(
    const float* __restrict__ x,     // (B,S)
    const float* __restrict__ hx1,   // (B,10) zeros, but honored
    const float* __restrict__ W_ih,  // (10,1)
    const float* __restrict__ b_ih,  // (10,)
    const float* __restrict__ W_hh,  // (10,10)
    const float* __restrict__ b_hh,  // (10,)
    const float* __restrict__ W1,    // (20,10)
    const float* __restrict__ W2,    // (2,20)
    const float* __restrict__ b2,    // (2,)
    float* __restrict__ out)         // (B,2)
{
    const int lane   = threadIdx.x & 31;
    const int wave   = threadIdx.x >> 5;
    const int tile   = blockIdx.x * TILES_PER_BLOCK + wave;
    const int batch0 = tile * 16;
    const bool lo    = lane < 16;
    const int n      = lane & 15;   // batch column within tile (both halves)
    const int m      = n;           // A-matrix row = lane%16

    // ---- A = W_hh as f16 16x32 (M=hid_out over lanes, K=hid_in), zero-padded.
    // A k order used here (contiguous, as B): lanes0-15: elem e <-> K=e (e<16).
    //                  lanes16-31: elem e <-> K=16+e (always 0).
    v16h A = {};
    {
        const int mm = (m < 10) ? m : 9;
        #pragma unroll
        for (int e = 0; e < 16; ++e) {
            int k  = lo ? e : (16 + e);
            int kk = (k < 10) ? k : 9;
            float w = W_hh[mm * 10 + kk];
            w = (m < 10 && k < 10) ? w : 0.0f;      // cndmask, no divergence
            A[e] = (_Float16)w;
        }
    }

    // ---- Per-lane C-matrix constants: C(j,n) = x[n,t]*w_in[j] + (b_ih+b_hh)[j].
    // C/D layout: vgpr r, lanes0-15 -> j=r ; lanes16-31 -> j=r+8. j>=10 -> 0.
    float wv[8], bv[8];
    #pragma unroll
    for (int r = 0; r < 8; ++r) {
        int j  = lo ? r : (r + 8);
        int jj = (j < 10) ? j : 9;
        float wval = W_ih[jj];                      // w_in = W_ih[:,0]
        float bval = b_ih[jj] + b_hh[jj];
        bool valid = (j < 10);
        wv[r] = valid ? wval : 0.0f;
        bv[r] = valid ? bval : 0.0f;
    }

    // ---- B = H^T as f16 32x16 (K=hid over vgprs/half-lanes, N=batch over lanes).
    // vgpr v lanes0-15 = rows K=2v,2v+1 packed; lanes16-31 = K=16+2v (always 0).
    union BU { v16h v; h2v p[8]; } bu;
    { v16h z = {}; bu.v = z; }                      // p[5..7] stay 0 forever
    const h2v z2 = {};
    {
        const float* hrow = hx1 + (size_t)(batch0 + n) * 10;
        #pragma unroll
        for (int kp = 0; kp < 5; ++kp) {
            h2v p;
            p.x = (_Float16)hrow[2 * kp];
            p.y = (_Float16)hrow[2 * kp + 1];
            bu.p[kp] = lo ? p : z2;
        }
    }

    // ---- Sequential scan: one WMMA per step computes all 16 batch rows.
    // Software-pipelined x stream: load quad for t+4 while computing t..t+3.
    const float* xrow = x + (size_t)(batch0 + n) * S_LEN;
    float T[8] = {};
    float s0 = 0.0f, s1 = 0.0f;

    float4 xq = *reinterpret_cast<const float4*>(xrow);

    for (int t = 0; t < S_LEN; t += 4) {
        // issue next quad load early (clamped on final iter; uniform select)
        const int tn = (t + 4 < S_LEN) ? (t + 4) : t;
        const float4 xq_next = *reinterpret_cast<const float4*>(xrow + tn);
        // stage each lane's private x stream ~16 iterations ahead
        __builtin_prefetch(xrow + t + 64, 0, 0);

        const float xs4[4] = {xq.x, xq.y, xq.z, xq.w};
        #pragma unroll
        for (int u = 0; u < 4; ++u) {
            const float xs = xs4[u];
            v8f c;
            #pragma unroll
            for (int r = 0; r < 8; ++r)
                c[r] = __builtin_fmaf(xs, wv[r], bv[r]);

            // D(j,n) = C(j,n) + sum_k W_hh[j,k] * h[n,k]
            v8f d = __builtin_amdgcn_wmma_f32_16x16x32_f16(
                false, A, false, bu.v, (short)0, c, false, false);

            #pragma unroll
            for (int r = 0; r < 8; ++r)
                T[r] = fast_tanh(d[r]);

            // rows j=8,9 live in lanes16-31 of vgpr0/1 -> bring to lanes0-15
            s0 = swap16(T[0]);
            s1 = swap16(T[1]);

            // repack next-step B (f16), zero lanes16-31 (those are K>=16 rows)
            #pragma unroll
            for (int kp = 0; kp < 4; ++kp) {
                h2v p = pack_f16(T[2 * kp], T[2 * kp + 1]);
                bu.p[kp] = lo ? p : z2;
            }
            {
                h2v p = pack_f16(s0, s1);
                bu.p[4] = lo ? p : z2;
            }
        }

        xq = xq_next;   // loadcnt wait lands here, after the 4-step compute
    }

    // ---- FC head: lanes 0-15 each hold h[0..9] for their batch row.
    if (lo) {
        float h[10];
        #pragma unroll
        for (int r = 0; r < 8; ++r) h[r] = T[r];
        h[8] = s0;
        h[9] = s1;

        float o0 = b2[0], o1 = b2[1];
        #pragma unroll
        for (int c2 = 0; c2 < 20; ++c2) {
            float f = 0.0f;
            #pragma unroll
            for (int j = 0; j < 10; ++j)
                f = __builtin_fmaf(h[j], W1[c2 * 10 + j], f);
            f = (f > 0.0f) ? f : 0.0f;              // relu
            o0 = __builtin_fmaf(W2[c2],      f, o0);
            o1 = __builtin_fmaf(W2[20 + c2], f, o1);
        }
        float* op = out + (size_t)(batch0 + n) * 2;
        op[0] = o0;
        op[1] = o1;
    }
}

extern "C" void kernel_launch(void* const* d_in, const int* in_sizes, int n_in,
                              void* d_out, int out_size, void* d_ws, size_t ws_size,
                              hipStream_t stream) {
    (void)in_sizes; (void)n_in; (void)out_size; (void)d_ws; (void)ws_size;
    const float* x    = (const float*)d_in[0];
    const float* hx1  = (const float*)d_in[1];
    const float* W_ih = (const float*)d_in[2];
    const float* b_ih = (const float*)d_in[3];
    const float* W_hh = (const float*)d_in[4];
    const float* b_hh = (const float*)d_in[5];
    const float* W1   = (const float*)d_in[6];
    const float* W2   = (const float*)d_in[7];
    const float* b2   = (const float*)d_in[8];
    float* out = (float*)d_out;

    const int n_tiles = B_TOT / 16;                      // 512 waves
    dim3 grid(n_tiles / TILES_PER_BLOCK);                // 128 blocks
    dim3 block(BLOCK_THREADS);                           // 128 threads (4 waves)
    rnn_wmma_kernel<<<grid, block, 0, stream>>>(
        x, hx1, W_ih, b_ih, W_hh, b_hh, W1, W2, b2, out);
}
